// FromToUpdater_23003844837554
// MI455X (gfx1250) — matrix operand order repaired and checked by reading, not compiled, not run on hardware
//
#include <hip/hip_runtime.h>
#include <hip/hip_bf16.h>

// Problem dimensions (from reference)
#define D_DIM       2048
#define H_DIM       4096
#define B_DIM       256
#define TOTAL_FROM  (D_DIM + H_DIM)           // 6144
#define TOTAL_ROWS  (TOTAL_FROM + H_DIM + 1)  // 10241
#define NPAD        10256                      // 641 * 16 (N padded to tile)
#define NTILES      641
#define USTRIDE     10244                      // U row stride (multiple of 4 floats -> 16B aligned rows)

typedef __attribute__((ext_vector_type(16))) __bf16          v16bf;
typedef __attribute__((ext_vector_type(8)))  __bf16          v8bf;
typedef __attribute__((ext_vector_type(8)))  float           v8f;
typedef __attribute__((ext_vector_type(8)))  unsigned short  v8us;

__device__ __forceinline__ unsigned short f32_to_bf16_rne(float f) {
    unsigned int u = __float_as_uint(f);
    unsigned int r = u + 0x7FFFu + ((u >> 16) & 1u);
    return (unsigned short)(r >> 16);
}

__device__ __forceinline__ float sigmoid_fast(float z) {
    return 1.0f / (1.0f + __expf(-z));
}

// ---------------------------------------------------------------------------
// Phase 0: f32 -> bf16 conversion of hyper_W (padded to NPAD rows) and X.
// ---------------------------------------------------------------------------
__global__ __launch_bounds__(256)
void conv_w_bf16(const float* __restrict__ W, unsigned short* __restrict__ Wbf) {
    size_t i = ((size_t)blockIdx.x * blockDim.x + threadIdx.x) * 8;
    if (i >= (size_t)NPAD * D_DIM) return;
    size_t row = i / D_DIM;              // 8 elems never cross a row (D%8==0)
    v8us o;
    if (row < (size_t)TOTAL_ROWS) {
        const float4* src = reinterpret_cast<const float4*>(W + i);
        float4 a = src[0], b = src[1];
        o[0] = f32_to_bf16_rne(a.x); o[1] = f32_to_bf16_rne(a.y);
        o[2] = f32_to_bf16_rne(a.z); o[3] = f32_to_bf16_rne(a.w);
        o[4] = f32_to_bf16_rne(b.x); o[5] = f32_to_bf16_rne(b.y);
        o[6] = f32_to_bf16_rne(b.z); o[7] = f32_to_bf16_rne(b.w);
    } else {
        o = (v8us)0;                     // zero-pad rows >= TOTAL_ROWS
    }
    *reinterpret_cast<v8us*>(Wbf + i) = o;
}

__global__ __launch_bounds__(256)
void conv_x_bf16(const float* __restrict__ X, unsigned short* __restrict__ Xbf) {
    size_t i = ((size_t)blockIdx.x * blockDim.x + threadIdx.x) * 8;
    if (i >= (size_t)B_DIM * D_DIM) return;
    const float4* src = reinterpret_cast<const float4*>(X + i);
    float4 a = src[0], b = src[1];
    v8us o;
    o[0] = f32_to_bf16_rne(a.x); o[1] = f32_to_bf16_rne(a.y);
    o[2] = f32_to_bf16_rne(a.z); o[3] = f32_to_bf16_rne(a.w);
    o[4] = f32_to_bf16_rne(b.x); o[5] = f32_to_bf16_rne(b.y);
    o[6] = f32_to_bf16_rne(b.z); o[7] = f32_to_bf16_rne(b.w);
    *reinterpret_cast<v8us*>(Xbf + i) = o;
}

// ---------------------------------------------------------------------------
// Phase 1: U[s][r] = sum_k X[s][k] * hyper_W[r][k] + b[r]
//   One GEMM: M=256 samples, N=10241(->10256) hyper rows, K=2048.
//   One 16x16 output tile per wave, K-loop in steps of 32 with
//   v_wmma_f32_16x16x32_bf16. 4 waves / block.
// ---------------------------------------------------------------------------
__global__ __launch_bounds__(128)
void hyper_gemm_wmma(const unsigned short* __restrict__ Xbf,
                     const unsigned short* __restrict__ Wbf,
                     const float* __restrict__ bias,
                     float* __restrict__ U) {
    const int wave = threadIdx.x >> 5;
    const int lane = threadIdx.x & 31;
    const int tile = blockIdx.x * 4 + wave;
    const int mt = tile & 15;     // sample tile   (16 tiles of 16)
    const int nt = tile >> 4;     // hyper-row tile (641 tiles of 16)
    const int hi = lane >> 4;     // lane half
    const int lo = lane & 15;

    // A fragment: row m = lo (samples). ISA layout => two contiguous 16B
    // chunks per lane at K offsets (hi*8) and (16 + hi*8).
    const unsigned short* arow = Xbf + (size_t)(mt * 16 + lo) * D_DIM;
    // B fragment: col n = lo (hyper rows). Same K layout as A: two 16B
    // chunks per lane at K offsets (hi*8) and (16 + hi*8).
    const unsigned short* brow = Wbf + (size_t)(nt * 16 + lo) * D_DIM;

    v8f c = {};
    for (int k0 = 0; k0 < D_DIM; k0 += 32) {
        v8bf a0 = *reinterpret_cast<const v8bf*>(arow + k0 + hi * 8);
        v8bf a1 = *reinterpret_cast<const v8bf*>(arow + k0 + 16 + hi * 8);
        v16bf a;
#pragma unroll
        for (int i = 0; i < 8; ++i) { a[i] = a0[i]; a[8 + i] = a1[i]; }
        v8bf b0 = *reinterpret_cast<const v8bf*>(brow + k0 + hi * 8);
        v8bf b1 = *reinterpret_cast<const v8bf*>(brow + k0 + 16 + hi * 8);
        v16bf b;
#pragma unroll
        for (int i = 0; i < 8; ++i) { b[i] = b0[i]; b[8 + i] = b1[i]; }
        c = __builtin_amdgcn_wmma_f32_16x16x32_bf16(
                /*neg_a=*/false, a, /*neg_b=*/false, b,
                /*c_mod=*/(short)0, c, /*reuse_a=*/false, /*reuse_b=*/false);
    }

    // Epilogue: C/D layout -> VGPR i holds M = i + hi*8, N = lo.
    const int n_global = nt * 16 + lo;
    if (n_global < TOTAL_ROWS) {
        const float bb = bias[n_global];
#pragma unroll
        for (int i = 0; i < 8; ++i) {
            const int m_global = mt * 16 + hi * 8 + i;
            U[(size_t)m_global * USTRIDE + n_global] = c[i] + bb;
        }
    }
}

// ---------------------------------------------------------------------------
// Phase 2a (per step s): fused W1 update + h = sigmoid(W1_new @ x_s).
//   One block per row r (4096 blocks). 256 threads x 8 columns.
//   W1 touched exactly once per step (read+write, L2 resident).
// ---------------------------------------------------------------------------
__global__ __launch_bounds__(256)
void step_w1_h(const float* __restrict__ x,   // [B][D]
               const float* __restrict__ U,   // [B][USTRIDE]
               float* __restrict__ w1,        // [H][D] carried state
               float* __restrict__ h,         // [H]
               int s) {
    __shared__ float red[256];
    const int r = blockIdx.x;
    const float* u  = U + (size_t)s * USTRIDE;
    const float  t1 = u[TOTAL_FROM + r];
    const float* xs = x + (size_t)s * D_DIM;
    float* wrow = w1 + (size_t)r * D_DIM;
    const int c0 = threadIdx.x * 8;

    float4 w0 = *reinterpret_cast<const float4*>(wrow + c0);
    float4 w1v = *reinterpret_cast<const float4*>(wrow + c0 + 4);
    float4 f0 = *reinterpret_cast<const float4*>(u + c0);       // f1 = u[:D]
    float4 f1 = *reinterpret_cast<const float4*>(u + c0 + 4);
    float4 x0 = *reinterpret_cast<const float4*>(xs + c0);
    float4 x1 = *reinterpret_cast<const float4*>(xs + c0 + 4);

    float acc = 0.0f;
    float4 n0, n1;
    n0.x = sigmoid_fast(10.0f * (w0.x + t1 * f0.x - 0.5f)); acc += n0.x * x0.x;
    n0.y = sigmoid_fast(10.0f * (w0.y + t1 * f0.y - 0.5f)); acc += n0.y * x0.y;
    n0.z = sigmoid_fast(10.0f * (w0.z + t1 * f0.z - 0.5f)); acc += n0.z * x0.z;
    n0.w = sigmoid_fast(10.0f * (w0.w + t1 * f0.w - 0.5f)); acc += n0.w * x0.w;
    n1.x = sigmoid_fast(10.0f * (w1v.x + t1 * f1.x - 0.5f)); acc += n1.x * x1.x;
    n1.y = sigmoid_fast(10.0f * (w1v.y + t1 * f1.y - 0.5f)); acc += n1.y * x1.y;
    n1.z = sigmoid_fast(10.0f * (w1v.z + t1 * f1.z - 0.5f)); acc += n1.z * x1.z;
    n1.w = sigmoid_fast(10.0f * (w1v.w + t1 * f1.w - 0.5f)); acc += n1.w * x1.w;

    *reinterpret_cast<float4*>(wrow + c0)     = n0;
    *reinterpret_cast<float4*>(wrow + c0 + 4) = n1;

    red[threadIdx.x] = acc;
    __syncthreads();
    for (int off = 128; off > 0; off >>= 1) {
        if (threadIdx.x < off) red[threadIdx.x] += red[threadIdx.x + off];
        __syncthreads();
    }
    if (threadIdx.x == 0) h[r] = sigmoid_fast(red[0]);
}

// ---------------------------------------------------------------------------
// Phase 2b (per step s): w2 update + y = sigmoid(w2_new . h). One block.
// ---------------------------------------------------------------------------
__global__ __launch_bounds__(256)
void step_w2_y(const float* __restrict__ U,
               const float* __restrict__ h,
               float* __restrict__ w2,        // [H] carried state
               float* __restrict__ out, int s) {
    __shared__ float red[256];
    const float* u  = U + (size_t)s * USTRIDE;
    const float  t2 = u[TOTAL_ROWS - 1];
    float acc = 0.0f;
    for (int j = threadIdx.x; j < H_DIM; j += 256) {
        const float f2 = u[D_DIM + j];                 // f2 = u[D:D+H]
        const float wn = sigmoid_fast(10.0f * (w2[j] + t2 * f2 - 0.5f));
        w2[j] = wn;
        acc += wn * h[j];
    }
    red[threadIdx.x] = acc;
    __syncthreads();
    for (int off = 128; off > 0; off >>= 1) {
        if (threadIdx.x < off) red[threadIdx.x] += red[threadIdx.x + off];
        __syncthreads();
    }
    if (threadIdx.x == 0) out[s] = sigmoid_fast(red[0]);
}

// ---------------------------------------------------------------------------
extern "C" void kernel_launch(void* const* d_in, const int* in_sizes, int n_in,
                              void* d_out, int out_size, void* d_ws, size_t ws_size,
                              hipStream_t stream) {
    const float* x      = (const float*)d_in[0];   // [B,1,D]
    const float* hyperW = (const float*)d_in[1];   // [TOTAL,D]
    const float* hyperb = (const float*)d_in[2];   // [TOTAL]
    const float* W1     = (const float*)d_in[3];   // [H,D]
    const float* W2     = (const float*)d_in[4];   // [1,H]
    float* out = (float*)d_out;                    // [B]

    char* ws = (char*)d_ws;
    size_t off = 0;
    auto alloc = [&](size_t bytes) -> void* {
        void* p = ws + off;
        off += (bytes + 255) & ~(size_t)255;
        return p;
    };
    unsigned short* Wbf = (unsigned short*)alloc((size_t)NPAD * D_DIM * 2); // 42 MB
    unsigned short* Xbf = (unsigned short*)alloc((size_t)B_DIM * D_DIM * 2);//  1 MB
    float* U   = (float*)alloc((size_t)B_DIM * USTRIDE * 4);                // 10.5 MB
    float* w1s = (float*)alloc((size_t)H_DIM * D_DIM * 4);                  // 33.5 MB
    float* w2s = (float*)alloc((size_t)H_DIM * 4);
    float* hbuf= (float*)alloc((size_t)H_DIM * 4);

    // Re-seed carried state each call (deterministic, capture-safe d2d copies).
    hipMemcpyAsync(w1s, W1, (size_t)H_DIM * D_DIM * 4, hipMemcpyDeviceToDevice, stream);
    hipMemcpyAsync(w2s, W2, (size_t)H_DIM * 4,         hipMemcpyDeviceToDevice, stream);

    // Phase 0: bf16 conversion (hyper_W read once, ever).
    conv_w_bf16<<<((size_t)NPAD * D_DIM / 8 + 255) / 256, 256, 0, stream>>>(hyperW, Wbf);
    conv_x_bf16<<<((size_t)B_DIM * D_DIM / 8 + 255) / 256, 256, 0, stream>>>(x, Xbf);

    // Phase 1: all 256 hyper mat-vecs as one WMMA GEMM.
    hyper_gemm_wmma<<<(16 * NTILES) / 4, 128, 0, stream>>>(Xbf, Wbf, hyperb, U);

    // Phase 2: sequential scan over samples (true data dependency).
    for (int s = 0; s < B_DIM; ++s) {
        step_w1_h<<<H_DIM, 256, 0, stream>>>(x, U, w1s, hbuf, s);
        step_w2_y<<<1, 256, 0, stream>>>(U, hbuf, w2s, out, s);
    }
}
